// TimeAwareAttention_30416958390796
// MI455X (gfx1250) — matrix operand order repaired and checked by reading, not compiled, not run on hardware
//
#include <hip/hip_runtime.h>
#include <hip/hip_bf16.h>

// ---------------------------------------------------------------------------
// TimeAwareAttention on gfx1250 (MI455X, wave32, WMMA 16x16x32 f16 / f32-acc)
//
//   B=2, S=2048, D=1024, H=16, HD=64
//   Pipeline: [pack x->f16, W->f16 transposed] -> [QKV gemm 128x128 tiles,
//   double-buffered async global->LDS staging] -> [flash attention with
//   multiplicative time decay] -> [output projection gemm].
//
// WMMA fragment maps used here (A/B in contiguous k order, not ISA table order; lane l, hh=l>>4, li=l&15):
//   A (16x32 f16):  row = li,  elem e -> K = e + 16*hh
//   B (32x16 f16):  col = li,  elem e -> K = e + 16*hh
//   C/D (16x16 f32): elem j -> row = j + 8*hh, col = li
// ---------------------------------------------------------------------------

typedef __attribute__((ext_vector_type(16))) _Float16 v16h;
typedef __attribute__((ext_vector_type(8)))  float    v8f;
typedef __attribute__((ext_vector_type(4)))  int      v4i;

#define NB   1024          // model dim D
#define SEQ  2048
#define NH   16
#define HD   64
#define NROW 4096          // B*S
#define LSTR 40            // LDS row stride in halfs (32 data + 8 pad; rows stay 16B aligned: 80B)

#ifndef __has_builtin
#define __has_builtin(x) 0
#endif
#if __has_builtin(__builtin_amdgcn_global_load_async_to_lds_b128)
#define USE_ASYNC_LDS 1
#else
#define USE_ASYNC_LDS 0
#endif

// Builtin signature (verified via hipcc diagnostic): param0 = v4i AS1*,
// param1 = v4i AS3*, then two immediate ints (offset, cpol).
typedef __attribute__((address_space(1))) v4i g_v4i;
typedef __attribute__((address_space(3))) v4i l_v4i;

__device__ __forceinline__ void async_b128(const void* g, void* l) {
#if USE_ASYNC_LDS
  // AS1 value == flat value for global; LDS aperture maps addr[31:0] -> LDS
  // offset (ISA 10.2), so integer round-trips are value-preserving here.
  __builtin_amdgcn_global_load_async_to_lds_b128(
      (g_v4i*)(unsigned long long)g,
      (l_v4i*)(unsigned)(unsigned long long)l,
      0, 0);
#else
  *(uint4*)l = *(const uint4*)g;   // falls back to global_load_b128 + ds_store_b128
#endif
}

__device__ __forceinline__ void wait_async0() {
#if USE_ASYNC_LDS
#if __has_builtin(__builtin_amdgcn_s_wait_asynccnt)
  __builtin_amdgcn_s_wait_asynccnt(0);
#else
  asm volatile("s_wait_asynccnt 0" ::: "memory");
#endif
#endif
}

// Partial drain: leave the 4 in-flight transfers of the *next* tile pending.
__device__ __forceinline__ void wait_async4() {
#if USE_ASYNC_LDS
#if __has_builtin(__builtin_amdgcn_s_wait_asynccnt)
  __builtin_amdgcn_s_wait_asynccnt(4);
#else
  asm volatile("s_wait_asynccnt 4" ::: "memory");
#endif
#endif
}

union FragU { unsigned u[8]; v16h v; };

__device__ __forceinline__ v16h load_frag_a(const unsigned* base32, int hh) {
  FragU f;
#pragma unroll
  for (int p = 0; p < 8; ++p) {
    const int offh = 2 * p + 16 * hh;  // even
    f.u[p] = base32[offh >> 1];
  }
  return f.v;
}

__device__ __forceinline__ v16h load_frag_b(const unsigned* base32, int hh) {
  FragU f;
#pragma unroll
  for (int p = 0; p < 8; ++p) f.u[p] = base32[p + 8 * hh];
  return f.v;
}

__device__ __forceinline__ v8f wmma_f16(v16h a, v16h b, v8f c) {
  return __builtin_amdgcn_wmma_f32_16x16x32_f16(false, a, false, b,
                                                (short)0, c, false, false);
}

// ---------------------------------------------------------------------------
// Kernel 0a: x f32 -> f16, row-major passthrough.
// ---------------------------------------------------------------------------
__global__ __launch_bounds__(256) void taa_pack_x(
    const float* __restrict__ x, _Float16* __restrict__ xh)
{
  const size_t i4 = (size_t)blockIdx.x * 256 + threadIdx.x;   // float4 index
  const float4 v = ((const float4*)x)[i4];
  union { _Float16 h[4]; uint2 u; } o;
  o.h[0] = (_Float16)v.x; o.h[1] = (_Float16)v.y;
  o.h[2] = (_Float16)v.z; o.h[3] = (_Float16)v.w;
  ((uint2*)xh)[i4] = o.u;
}

// ---------------------------------------------------------------------------
// Kernel 0b: W[k][n] f32 -> WT[n][k] f16 (LDS-tiled 32x32 transpose).
// z selects Wq/Wk/Wv/Wo.
// ---------------------------------------------------------------------------
__global__ __launch_bounds__(256) void taa_pack_wT(
    const float* __restrict__ Wq, const float* __restrict__ Wk,
    const float* __restrict__ Wv, const float* __restrict__ Wo,
    _Float16* __restrict__ wT)    // 4 matrices of 1024*1024, contiguous
{
  __shared__ _Float16 t[32][33];
  const int tid = threadIdx.x;
  const int k0 = blockIdx.x * 32, n0 = blockIdx.y * 32, z = blockIdx.z;
  const float* W = (z == 0) ? Wq : (z == 1) ? Wk : (z == 2) ? Wv : Wo;
  _Float16* dst = wT + (size_t)z * NB * NB;
  const int r = tid >> 5, c = tid & 31;
#pragma unroll
  for (int i = 0; i < 4; ++i) {
    const int rr = r + i * 8;
    t[rr][c] = (_Float16)W[(size_t)(k0 + rr) * NB + n0 + c];
  }
  __syncthreads();
#pragma unroll
  for (int i = 0; i < 4; ++i) {
    const int rr = r + i * 8;
    dst[(size_t)(n0 + rr) * NB + k0 + c] = t[c][rr];
  }
}

// ---------------------------------------------------------------------------
// Shared GEMM mainloop: C[128x128] += A[128xK] * BT[128xK]^T, K-step 32,
// double-buffered async staging.  8 waves; wave tile 32x64.
// acc[mi][n]: mi in {0,1} (16-row strips), n in 0..3 (16-col strips).
// ---------------------------------------------------------------------------
__device__ __forceinline__ void gemm_mainloop_128(
    const _Float16* __restrict__ A, const _Float16* __restrict__ BT,
    _Float16* As0, _Float16* Bs0, _Float16* As1, _Float16* Bs1,
    int m0, int n0, int tid, v8f acc[2][4])
{
  const int wave = tid >> 5, l = tid & 31, hh = l >> 4, li = l & 15;
  const int msub = (wave & 3) * 32, nsub = (wave >> 2) * 64;

  // Stage one 128x32-half tile pair: 512 x 16B segments per tile, 2/thread,
  // 4 async instructions per wave per call.
  auto stage = [&](int k0, _Float16* As, _Float16* Bs) {
#pragma unroll
    for (int i = 0; i < 2; ++i) {
      const int e = tid + i * 256, r = e >> 2, seg = e & 3;
      async_b128(A  + (size_t)(m0 + r) * NB + k0 + seg * 8,
                 As + r * LSTR + seg * 8);
      async_b128(BT + (size_t)(n0 + r) * NB + k0 + seg * 8,
                 Bs + r * LSTR + seg * 8);
    }
  };

  stage(0, As0, Bs0);
  for (int k0 = 0; k0 < NB; k0 += 32) {
    _Float16* As = (k0 & 32) ? As1 : As0;
    _Float16* Bs = (k0 & 32) ? Bs1 : Bs0;
    if (k0 + 32 < NB) {
      // Prefetch next tile into the other buffer (its previous readers
      // finished before last iteration's trailing barrier), then drain only
      // the current tile's 4 transfers.
      stage(k0 + 32, (k0 & 32) ? As0 : As1, (k0 & 32) ? Bs0 : Bs1);
      wait_async4();
    } else {
      wait_async0();
    }
    __syncthreads();

    const v16h a0 = load_frag_a((const unsigned*)(As + (msub + li) * LSTR), hh);
    const v16h a1 = load_frag_a((const unsigned*)(As + (msub + 16 + li) * LSTR), hh);
    v16h b[4];
#pragma unroll
    for (int n = 0; n < 4; ++n)
      b[n] = load_frag_b((const unsigned*)(Bs + (nsub + n * 16 + li) * LSTR), hh);
#pragma unroll
    for (int n = 0; n < 4; ++n) {
      acc[0][n] = wmma_f16(a0, b[n], acc[0][n]);
      acc[1][n] = wmma_f16(a1, b[n], acc[1][n]);
    }
    __syncthreads();   // all reads done before this buffer is re-staged
  }
}

// ---------------------------------------------------------------------------
// Kernel 1: QKV projection from xh f16 and transposed f16 weights.
//   z==0 -> Q head-major [B,H,S,HD]; z==1 -> K head-major;
//   z==2 -> V transposed [B,H,HD,S] (b128-packed epilogue stores).
// ---------------------------------------------------------------------------
__global__ __launch_bounds__(256) void taa_qkv_gemm(
    const _Float16* __restrict__ xh, const _Float16* __restrict__ wT,
    const float* __restrict__ bq, const float* __restrict__ bk,
    const float* __restrict__ bv,
    _Float16* __restrict__ qbuf, _Float16* __restrict__ kbuf,
    _Float16* __restrict__ vbuf)
{
  __shared__ __align__(16) _Float16 As0[128 * LSTR];
  __shared__ __align__(16) _Float16 Bs0[128 * LSTR];
  __shared__ __align__(16) _Float16 As1[128 * LSTR];
  __shared__ __align__(16) _Float16 Bs1[128 * LSTR];
  const int tid = threadIdx.x;
  const int wave = tid >> 5, l = tid & 31, hh = l >> 4, li = l & 15;
  const int m0 = blockIdx.y * 128, n0 = blockIdx.x * 128;
  const int which = blockIdx.z;
  const _Float16* BT = wT + (size_t)which * NB * NB;
  const float* bias = (which == 0) ? bq : (which == 1) ? bk : bv;

  v8f acc[2][4] = {};
  gemm_mainloop_128(xh, BT, As0, Bs0, As1, Bs1, m0, n0, tid, acc);

  const int msub = (wave & 3) * 32, nsub = (wave >> 2) * 64;
#pragma unroll
  for (int mi = 0; mi < 2; ++mi)
#pragma unroll
    for (int n = 0; n < 4; ++n) {
      const int col  = n0 + nsub + n * 16 + li;
      const int head = col >> 6, hd = col & 63;
      const float bb = bias[col];
      const int row0 = m0 + msub + mi * 16 + 8 * hh;   // j=0 row
      const int b = row0 >> 11, s0 = row0 & (SEQ - 1);
      if (which == 2) {
        // 8 consecutive s at fixed hd -> one 16B store (s0 % 8 == 0).
        union { _Float16 h[8]; uint4 u; } pk;
#pragma unroll
        for (int j = 0; j < 8; ++j) pk.h[j] = (_Float16)(acc[mi][n][j] + bb);
        *(uint4*)(vbuf + (((size_t)(b * NH + head)) * HD + hd) * SEQ + s0) = pk.u;
      } else {
        _Float16* dst = (which == 0) ? qbuf : kbuf;
#pragma unroll
        for (int j = 0; j < 8; ++j)
          dst[(((size_t)(b * NH + head)) * SEQ + s0 + j) * HD + hd] =
              (_Float16)(acc[mi][n][j] + bb);
      }
    }
}

// ---------------------------------------------------------------------------
// Kernel 2: flash attention with multiplicative decay exp(-td*|tq-tk|).
// 4 independent waves per block; each wave owns 16 query rows of one (b,h),
// streams keys in tiles of 32.  Per tile: 4 WMMA (QK^T) + 4 WMMA (PV).
// ---------------------------------------------------------------------------
__global__ __launch_bounds__(128) void taa_attention(
    const _Float16* __restrict__ qh,   // [B,H,S,HD]
    const _Float16* __restrict__ kh,   // [B,H,S,HD]
    const _Float16* __restrict__ vh,   // [B,H,HD,S]
    const float* __restrict__ ts,      // [B,S]
    const float* __restrict__ time_decay,
    _Float16* __restrict__ att)        // [B,S,D]
{
  constexpr int PSTR = 34;
  __shared__ __align__(16) _Float16 pt[4][16 * PSTR];

  const int tid  = threadIdx.x;
  const int wave = tid >> 5, l = tid & 31, hh = l >> 4, li = l & 15;
  const int h  = blockIdx.y, b = blockIdx.z;
  const int bh = b * NH + h;
  const int q0 = blockIdx.x * 64 + wave * 16;
  const float td = time_decay[0];

  const unsigned* qrow =
      (const unsigned*)(qh + ((size_t)bh * SEQ + q0 + li) * HD);
  const v16h qa0 = load_frag_a(qrow, hh);
  const v16h qa1 = load_frag_a(qrow + 16, hh);

  float tq[8], m[8], lsum[8];
#pragma unroll
  for (int j = 0; j < 8; ++j) {
    tq[j]   = ts[b * SEQ + q0 + j + 8 * hh];
    m[j]    = -1e30f;
    lsum[j] = 0.0f;
  }
  v8f o[4] = {};

  for (int k0 = 0; k0 < SEQ; k0 += 32) {
    float pm[2][8];
#pragma unroll
    for (int a = 0; a < 2; ++a) {
      const int key = k0 + a * 16 + li;
      const unsigned* krow =
          (const unsigned*)(kh + ((size_t)bh * SEQ + key) * HD);
      v8f sc = {};
      sc = wmma_f16(qa0, load_frag_b(krow, hh), sc);
      sc = wmma_f16(qa1, load_frag_b(krow + 16, hh), sc);
      const float tk = ts[b * SEQ + key];
#pragma unroll
      for (int j = 0; j < 8; ++j)
        pm[a][j] = sc[j] * 0.125f * __expf(-td * fabsf(tq[j] - tk));
    }

#pragma unroll
    for (int j = 0; j < 8; ++j) {
      float mx = fmaxf(pm[0][j], pm[1][j]);
      mx = fmaxf(mx, __shfl_xor(mx, 1, 32));
      mx = fmaxf(mx, __shfl_xor(mx, 2, 32));
      mx = fmaxf(mx, __shfl_xor(mx, 4, 32));
      mx = fmaxf(mx, __shfl_xor(mx, 8, 32));
      const float mnew = fmaxf(m[j], mx);
      const float corr = __expf(m[j] - mnew);
      pm[0][j] = __expf(pm[0][j] - mnew);
      pm[1][j] = __expf(pm[1][j] - mnew);
      float rs = pm[0][j] + pm[1][j];
      rs += __shfl_xor(rs, 1, 32);
      rs += __shfl_xor(rs, 2, 32);
      rs += __shfl_xor(rs, 4, 32);
      rs += __shfl_xor(rs, 8, 32);
      lsum[j] = lsum[j] * corr + rs;
      m[j] = mnew;
      o[0][j] *= corr; o[1][j] *= corr; o[2][j] *= corr; o[3][j] *= corr;
    }

    _Float16* ptw = pt[wave];
#pragma unroll
    for (int a = 0; a < 2; ++a)
#pragma unroll
      for (int j = 0; j < 8; ++j)
        ptw[(j + 8 * hh) * PSTR + a * 16 + li] = (_Float16)pm[a][j];
    __syncthreads();

    const v16h pa = load_frag_a((const unsigned*)(ptw + li * PSTR), hh);
#pragma unroll
    for (int a4 = 0; a4 < 4; ++a4) {
      const unsigned* vrow = (const unsigned*)(
          vh + ((size_t)bh * HD + a4 * 16 + li) * SEQ + k0);
      o[a4] = wmma_f16(pa, load_frag_b(vrow, hh), o[a4]);
    }
    __syncthreads();
  }

  float inv[8];
#pragma unroll
  for (int j = 0; j < 8; ++j) inv[j] = (lsum[j] > 0.0f) ? 1.0f / lsum[j] : 0.0f;
#pragma unroll
  for (int a4 = 0; a4 < 4; ++a4)
#pragma unroll
    for (int j = 0; j < 8; ++j) {
      const int row = q0 + j + 8 * hh;
      const int col = h * HD + a4 * 16 + li;
      att[((size_t)b * SEQ + row) * NB + col] = (_Float16)(o[a4][j] * inv[j]);
    }
}

// ---------------------------------------------------------------------------
// Kernel 3: output projection.  out f32 = att f16 @ WoT^T + bo.
// ---------------------------------------------------------------------------
__global__ __launch_bounds__(256) void taa_out_gemm(
    const _Float16* __restrict__ att, const _Float16* __restrict__ woT,
    const float* __restrict__ bo, float* __restrict__ out)
{
  __shared__ __align__(16) _Float16 As0[128 * LSTR];
  __shared__ __align__(16) _Float16 Bs0[128 * LSTR];
  __shared__ __align__(16) _Float16 As1[128 * LSTR];
  __shared__ __align__(16) _Float16 Bs1[128 * LSTR];
  const int tid = threadIdx.x;
  const int wave = tid >> 5, l = tid & 31, hh = l >> 4, li = l & 15;
  const int m0 = blockIdx.y * 128, n0 = blockIdx.x * 128;

  v8f acc[2][4] = {};
  gemm_mainloop_128(att, woT, As0, Bs0, As1, Bs1, m0, n0, tid, acc);

  const int msub = (wave & 3) * 32, nsub = (wave >> 2) * 64;
#pragma unroll
  for (int mi = 0; mi < 2; ++mi)
#pragma unroll
    for (int n = 0; n < 4; ++n) {
      const int col = n0 + nsub + n * 16 + li;
      const float bb = bo[col];
#pragma unroll
      for (int j = 0; j < 8; ++j) {
        const int row = m0 + msub + mi * 16 + j + 8 * hh;
        out[(size_t)row * NB + col] = acc[mi][n][j] + bb;
      }
    }
}

// ---------------------------------------------------------------------------
extern "C" void kernel_launch(void* const* d_in, const int* in_sizes, int n_in,
                              void* d_out, int out_size, void* d_ws, size_t ws_size,
                              hipStream_t stream) {
  const float* x  = (const float*)d_in[0];
  const float* ts = (const float*)d_in[1];
  const float* Wq = (const float*)d_in[2];  const float* bq = (const float*)d_in[3];
  const float* Wk = (const float*)d_in[4];  const float* bk = (const float*)d_in[5];
  const float* Wv = (const float*)d_in[6];  const float* bv = (const float*)d_in[7];
  const float* Wo = (const float*)d_in[8];  const float* bo = (const float*)d_in[9];
  const float* td = (const float*)d_in[10];
  float* out = (float*)d_out;

  // Workspace (f16 elements): q,k,v,att (4M each) + xh (4M) + 4 transposed W (1M each) = 24M halfs = 48MB.
  _Float16* qbuf = (_Float16*)d_ws;
  _Float16* kbuf = qbuf + (size_t)NROW * NB;
  _Float16* vbuf = kbuf + (size_t)NROW * NB;
  _Float16* attb = vbuf + (size_t)NROW * NB;
  _Float16* xh   = attb + (size_t)NROW * NB;
  _Float16* wT   = xh   + (size_t)NROW * NB;   // [4][1024][1024], order q,k,v,o

  taa_pack_x<<<dim3((NROW * NB) / (256 * 4)), 256, 0, stream>>>(x, xh);
  taa_pack_wT<<<dim3(NB / 32, NB / 32, 4), 256, 0, stream>>>(Wq, Wk, Wv, Wo, wT);
  taa_qkv_gemm<<<dim3(NB / 128, NROW / 128, 3), 256, 0, stream>>>(
      xh, wT, bq, bk, bv, qbuf, kbuf, vbuf);
  taa_attention<<<dim3(SEQ / 64, NH, 2), 128, 0, stream>>>(
      qbuf, kbuf, vbuf, ts, td, attb);
  taa_out_gemm<<<dim3(NB / 128, NROW / 128), 256, 0, stream>>>(
      attb, wT + (size_t)3 * NB * NB, bo, out);
}
